// LSTMCellWithLNAndDropout_60550448939164
// MI455X (gfx1250) — matrix operand order repaired and checked by reading, not compiled, not run on hardware
//
#include <hip/hip_runtime.h>
#include <hip/hip_bf16.h>

typedef __attribute__((ext_vector_type(16))) _Float16 v16h;
typedef __attribute__((ext_vector_type(8)))  _Float16 v8h;
typedef __attribute__((ext_vector_type(8)))  float    v8f;

#define BB   4096          // batch
#define DD   1024          // input dim
#define HH   1024          // hidden dim
#define KK   2048          // D + H
#define NN   4096          // 4H
#define LN_EPS 1e-5f

// ---------------------------------------------------------------------------
// Kernel 1: pack A = [x | prev_h] -> f16, row-major [BB x KK]
// ---------------------------------------------------------------------------
__global__ __launch_bounds__(256) void pack_A_kernel(
    const float* __restrict__ x, const float* __restrict__ prev_h,
    _Float16* __restrict__ Af)
{
    const size_t total  = (size_t)BB * KK;
    const size_t stride = (size_t)gridDim.x * blockDim.x;
    for (size_t idx = (size_t)blockIdx.x * blockDim.x + threadIdx.x;
         idx < total; idx += stride) {
        const int m = (int)(idx >> 11);          // / KK
        const int k = (int)(idx & (KK - 1));
        float v = (k < DD) ? x[(size_t)m * DD + k]
                           : prev_h[(size_t)m * HH + (k - DD)];
        Af[idx] = (_Float16)v;
    }
}

// ---------------------------------------------------------------------------
// Kernel 2: Wt[n][k] = Wcat[k][n] -> f16, via LDS tile transpose.
// Wcat[k][n] = (k < DD) ? Wx[k][n] : Wh[k-DD][n]
// ---------------------------------------------------------------------------
__global__ __launch_bounds__(256) void pack_Wt_kernel(
    const float* __restrict__ Wx, const float* __restrict__ Wh,
    _Float16* __restrict__ Wt)
{
    __shared__ float tile[32][33];
    const int nBase = blockIdx.x * 32;           // N tiles: 4096/32 = 128
    const int kBase = blockIdx.y * 32;           // K tiles: 2048/32 = 64
    const int tx = threadIdx.x;                  // 0..31
    const int ty = threadIdx.y;                  // 0..7

    #pragma unroll
    for (int r = ty; r < 32; r += 8) {
        const int k = kBase + r;
        const int n = nBase + tx;
        float v = (k < DD) ? Wx[(size_t)k * NN + n]
                           : Wh[(size_t)(k - DD) * NN + n];
        tile[r][tx] = v;
    }
    __syncthreads();
    #pragma unroll
    for (int r = ty; r < 32; r += 8) {
        const int n = nBase + r;
        const int k = kBase + tx;
        Wt[(size_t)n * KK + k] = (_Float16)tile[tx][r];
    }
}

// ---------------------------------------------------------------------------
// Kernel 3: GEMM  Aacc[M=4096][N=4096] = Af[M][K] * Wt[N][K]^T  (f16 in, f32 acc)
// 256 threads = 8 waves; wave tile 64(M) x 32(N); block tile 128 x 128.
// Software-pipelined: two fragment register sets, K unrolled by 64.
// ---------------------------------------------------------------------------
__device__ __forceinline__ void load_frags(
    const _Float16* const* aPtr, const _Float16* const* bPtr,
    int kk, v16h* af, v16h* bf)
{
    #pragma unroll
    for (int i = 0; i < 4; ++i) {
        union { v16h v; v8h h[2]; } u;
        u.h[0] = *(const v8h*)(aPtr[i] + kk);
        u.h[1] = *(const v8h*)(aPtr[i] + kk + 16);
        af[i] = u.v;
    }
    #pragma unroll
    for (int j = 0; j < 2; ++j) {
        union { v16h v; v8h h[2]; } u;
        u.h[0] = *(const v8h*)(bPtr[j] + kk);
        u.h[1] = *(const v8h*)(bPtr[j] + kk + 16);
        bf[j] = u.v;
    }
}

__device__ __forceinline__ void wmma8(const v16h* af, const v16h* bf,
                                      v8f acc[4][2])
{
    #pragma unroll
    for (int i = 0; i < 4; ++i)
        #pragma unroll
        for (int j = 0; j < 2; ++j)
            acc[i][j] = __builtin_amdgcn_wmma_f32_16x16x32_f16(
                /*neg_a=*/false, af[i], /*neg_b=*/false, bf[j],
                /*c_mod=*/(short)0, acc[i][j],
                /*reuse_a=*/false, /*reuse_b=*/false);
}

__global__ __launch_bounds__(256) void gemm_wmma_kernel(
    const _Float16* __restrict__ A, const _Float16* __restrict__ Bt,
    float* __restrict__ Cout)
{
    const int lane = threadIdx.x & 31;
    const int wave = threadIdx.x >> 5;
    const int wm   = wave & 1;                   // 2 waves along M
    const int wn   = wave >> 1;                  // 4 waves along N
    const int mBase = blockIdx.y * 128 + wm * 64;
    const int nBase = blockIdx.x * 128 + wn * 32;

    // Per-lane fragment addressing (CDNA5 16-bit WMMA layouts):
    //  A (16x32): lane L holds row L&15, K = k0..k0+7 and k0+16..k0+23, k0 = 8*(L>=16)
    //  B (32x16): lane L holds col L&15, K = k1..k1+7 and k1+16..k1+23, k1 = 8*(L>=16)
    const int aRow  = lane & 15;
    const int aKoff = (lane >> 4) << 3;          // 0 or 8
    const int bCol  = lane & 15;
    const int bKoff = (lane >> 4) << 3;          // 0 or 8

    const _Float16* aPtr[4];
    #pragma unroll
    for (int i = 0; i < 4; ++i)
        aPtr[i] = A + (size_t)(mBase + i * 16 + aRow) * KK + aKoff;
    const _Float16* bPtr[2];
    #pragma unroll
    for (int j = 0; j < 2; ++j)
        bPtr[j] = Bt + (size_t)(nBase + j * 16 + bCol) * KK + bKoff;

    v8f acc[4][2] = {};

    v16h afA[4], bfA[2], afB[4], bfB[2];
    load_frags(aPtr, bPtr, 0, afA, bfA);

    for (int kk = 0; kk < KK; kk += 64) {
        // stage B fragments for kk+32 while WMMAs on kk run
        load_frags(aPtr, bPtr, kk + 32, afB, bfB);

        // far prefetch into near caches (speculative; OOB drops silently)
        __builtin_prefetch(aPtr[0] + kk + 512, 0, 3);
        __builtin_prefetch(aPtr[2] + kk + 512, 0, 3);
        __builtin_prefetch(bPtr[0] + kk + 512, 0, 3);

        wmma8(afA, bfA, acc);

        if (kk + 64 < KK)
            load_frags(aPtr, bPtr, kk + 64, afA, bfA);

        wmma8(afB, bfB, acc);
    }

    // C/D layout: VGPR v -> row v + 8*(lane>=16), col = lane&15
    const int cRowOff = (lane >> 4) << 3;
    const int cCol    = lane & 15;
    #pragma unroll
    for (int i = 0; i < 4; ++i)
        #pragma unroll
        for (int j = 0; j < 2; ++j) {
            float* base = Cout + (size_t)(mBase + i * 16 + cRowOff) * NN
                               + (nBase + j * 16 + cCol);
            #pragma unroll
            for (int v = 0; v < 8; ++v)
                base[(size_t)v * NN] = acc[i][j][v];
        }
}

// ---------------------------------------------------------------------------
// Kernel 4: per-row LayerNorm + LSTM gates. One 256-thread block per row.
// ---------------------------------------------------------------------------
__global__ __launch_bounds__(256) void ln_gates_kernel(
    const float* __restrict__ Aacc, const float* __restrict__ bvec,
    const float* __restrict__ gamma, const float* __restrict__ beta,
    const float* __restrict__ prev_c,
    float* __restrict__ out_h, float* __restrict__ out_c)
{
    __shared__ float row[NN];
    __shared__ float red[512];
    __shared__ float mu_s, rstd_s;

    const int r = blockIdx.x;
    const int t = threadIdx.x;
    const float* src = Aacc + (size_t)r * NN;

    float s = 0.f, s2 = 0.f;
    #pragma unroll
    for (int u = 0; u < 16; ++u) {
        const int idx = t + 256 * u;
        float v = src[idx] + bvec[idx];
        row[idx] = v;
        s  += v;
        s2 += v * v;
    }
    red[t] = s;
    red[t + 256] = s2;
    __syncthreads();
    for (int off = 128; off > 0; off >>= 1) {
        if (t < off) {
            red[t]       += red[t + off];
            red[t + 256] += red[t + 256 + off];
        }
        __syncthreads();
    }
    if (t == 0) {
        const float inv = 1.f / (float)NN;
        float mu  = red[0] * inv;
        float var = red[256] * inv - mu * mu;
        mu_s   = mu;
        rstd_s = rsqrtf(var + LN_EPS);
    }
    __syncthreads();
    const float mu = mu_s, rstd = rstd_s;

    #pragma unroll
    for (int u = 0; u < 4; ++u) {
        const int j = t + 256 * u;               // 0..1023
        float ai = (row[j]        - mu) * rstd * gamma[j]        + beta[j];
        float af = (row[j + 1024] - mu) * rstd * gamma[j + 1024] + beta[j + 1024];
        float ao = (row[j + 2048] - mu) * rstd * gamma[j + 2048] + beta[j + 2048];
        float ag = (row[j + 3072] - mu) * rstd * gamma[j + 3072] + beta[j + 3072];
        float gi = 1.f / (1.f + __expf(-ai));
        float gf = 1.f / (1.f + __expf(-af));
        float go = 1.f / (1.f + __expf(-ao));
        float gg = tanhf(ag);
        const size_t o = (size_t)r * HH + j;
        float c = gf * prev_c[o] + gi * gg;
        float h = go * tanhf(c);
        out_h[o] = h;
        out_c[o] = c;
    }
}

// ---------------------------------------------------------------------------
// Launcher
// ---------------------------------------------------------------------------
extern "C" void kernel_launch(void* const* d_in, const int* in_sizes, int n_in,
                              void* d_out, int out_size, void* d_ws, size_t ws_size,
                              hipStream_t stream) {
    const float* x      = (const float*)d_in[0];
    const float* prev_h = (const float*)d_in[1];
    const float* prev_c = (const float*)d_in[2];
    const float* Wx     = (const float*)d_in[3];
    const float* Wh     = (const float*)d_in[4];
    const float* bvec   = (const float*)d_in[5];
    const float* gamma  = (const float*)d_in[6];
    const float* beta   = (const float*)d_in[7];

    float* out   = (float*)d_out;
    float* out_h = out;                              // [B, H]
    float* out_c = out + (size_t)BB * HH;            // [B, H]

    char* ws = (char*)d_ws;
    _Float16* Af   = (_Float16*)ws;                  // 16 MB: [B x K] f16
    _Float16* Wt   = (_Float16*)(ws + (16u << 20));  // 16 MB: [N x K] f16
    float*    Aacc = (float*)  (ws + (32u << 20));   // 64 MB: [B x N] f32

    pack_A_kernel<<<dim3(8192), dim3(256), 0, stream>>>(x, prev_h, Af);
    pack_Wt_kernel<<<dim3(NN / 32, KK / 32), dim3(32, 8), 0, stream>>>(Wx, Wh, Wt);
    gemm_wmma_kernel<<<dim3(NN / 128, BB / 128), dim3(256), 0, stream>>>(Af, Wt, Aacc);
    ln_gates_kernel<<<dim3(BB), dim3(256), 0, stream>>>(Aacc, bvec, gamma, beta,
                                                        prev_c, out_h, out_c);
}
